// DCNv1Module_82188494176854
// MI455X (gfx1250) — matrix operand order repaired and checked by reading, not compiled, not run on hardware
//
#include <hip/hip_runtime.h>
#include <hip/hip_bf16.h>

typedef __attribute__((ext_vector_type(16))) _Float16 v16h;
typedef __attribute__((ext_vector_type(8)))  float    v8f;

// Problem constants
#define BB   4
#define CC   64
#define OO   64
#define HH   128
#define WW   128
#define HW   16384          // H*W
#define K2   9
#define KD   576            // C*9
#define NPIX 65536          // B*H*W

// Shared GEMM tiling
#define CHUNK_K  288        // K rows per chunk (32 channels * 9 taps) -> 9 wmma steps
#define CHUNK_P  144        // packed f16 pair-rows per chunk
#define NT       64         // pixels per block (row-aligned chunk; 64 | W=128)
#define PSTRIDE  66         // u32 stride: 8*66 mod 64 == 16 -> conflict-free halves

// ---------------------------------------------------------------------------
// 1) Convert both weight tensors to f16 (contiguous pairs == packed K pairs).
//    w_dcn: [64][576]; w_off: [18][576] zero-padded to [32][576].
// ---------------------------------------------------------------------------
__global__ __launch_bounds__(256) void cvt_weights(const float* __restrict__ wd,
                                                   const float* __restrict__ wo,
                                                   unsigned short* __restrict__ od,
                                                   unsigned short* __restrict__ oo) {
  int i = blockIdx.x * 256 + threadIdx.x;
  if (i < OO * KD) {
    _Float16 h = (_Float16)wd[i];
    od[i] = __builtin_bit_cast(unsigned short, h);
  }
  if (i < 32 * KD) {
    int r = i / KD;
    float v = (r < 18) ? wo[i] : 0.0f;
    _Float16 h = (_Float16)v;
    oo[i] = __builtin_bit_cast(unsigned short, h);
  }
}

// ---------------------------------------------------------------------------
// 2) Offset conv as WMMA GEMM: M=32 (18 real ch), K=576, N=64 px per block.
//    8 waves = 2 o-tiles x 4 n-tiles. im2col patch (zero-pad) packed f16 in LDS.
// ---------------------------------------------------------------------------
__global__ __launch_bounds__(256) void offset_conv_wmma(const float*    __restrict__ x,
                                                        const unsigned* __restrict__ wpk, // f16[32][576] u32 pairs
                                                        const float*    __restrict__ bo,
                                                        float*          __restrict__ off) {
  __shared__ unsigned Ppk[CHUNK_P * PSTRIDE];

  const int tid   = threadIdx.x;
  const int n0    = blockIdx.x * NT;
  const int b     = n0 >> 14;
  const int rem   = n0 & (HW - 1);
  const int y     = rem >> 7;
  const int x0col = rem & (WW - 1);

  const int wave  = tid >> 5;
  const int lane  = tid & 31;
  const int otile = wave >> 2;     // 0..1
  const int ntile = wave & 3;      // 0..3
  const int h     = lane >> 4;
  const int l15   = lane & 15;

  v8f cacc = {};

  for (int cc = 0; cc < 2; ++cc) {
    __syncthreads();
    // im2col chunk: 144 pair-rows x 64 cols, both halves packed per thread
    for (int e = tid; e < CHUNK_P * NT; e += 256) {
      int pr = e >> 6;
      int j  = e & 63;
      unsigned pk = 0u;
      #pragma unroll
      for (int hv = 0; hv < 2; ++hv) {
        int Kg = cc * CHUNK_K + 2 * pr + hv;
        int c  = Kg / 9;
        int k  = Kg - 9 * c;
        int iy = y + k / 3 - 1;
        int ix = x0col + j + (k % 3) - 1;
        bool ok = (iy >= 0) && (iy < HH) && (ix >= 0) && (ix < WW);
        float s = ok ? x[((size_t)(b * CC + c) << 14) + (iy << 7) + ix] : 0.0f;
        _Float16 hf = (_Float16)s;
        pk |= ((unsigned)__builtin_bit_cast(unsigned short, hf)) << (16 * hv);
      }
      Ppk[pr * PSTRIDE + j] = pk;
    }
    __syncthreads();

    const unsigned* wbase = wpk + (size_t)(otile * 16 + l15) * (KD / 2) + cc * CHUNK_P;
    for (int kk = 0; kk < CHUNK_P; kk += 16) {
      union { v16h v; unsigned u[8]; } A, Bm;
      #pragma unroll
      for (int r = 0; r < 8; ++r) {
        int apair = kk + h * 8 + r;
        A.u[r] = wbase[apair];
        int bpair = kk + h * 8 + r;
        Bm.u[r] = Ppk[bpair * PSTRIDE + ntile * 16 + l15];
      }
      cacc = __builtin_amdgcn_wmma_f32_16x16x32_f16(
          false, A.v, false, Bm.v, (short)0, cacc, false, false);
    }
  }

  // Epilogue: only channels 0..17 are real
  const int px = x0col + ntile * 16 + l15;
  #pragma unroll
  for (int r = 0; r < 8; ++r) {
    int o = otile * 16 + h * 8 + r;
    if (o < 18)
      off[((size_t)(b * 18 + o) << 14) + (y << 7) + px] = cacc[r] + bo[o];
  }
}

// ---------------------------------------------------------------------------
// 3) Fused deformable-sample (fp32 bilinear) + WMMA GEMM (f16 -> f32 acc)
//    M=64, K=576, N=64 px per block. 8 waves = 4 o-tiles x 2 n-pairs;
//    each wave does 2 n-tiles sharing one A fragment (2x A reuse).
// ---------------------------------------------------------------------------
__global__ __launch_bounds__(256) void dcn_main(const float*    __restrict__ x,
                                                const float*    __restrict__ off,
                                                const unsigned* __restrict__ wpk,  // f16[64][576] u32 pairs
                                                const float*    __restrict__ bd,
                                                float*          __restrict__ out) {
  __shared__ unsigned Ppk[CHUNK_P * PSTRIDE];
  __shared__ int   sY0[K2 * NT];
  __shared__ int   sX0[K2 * NT];
  __shared__ float sWY[K2 * NT];
  __shared__ float sWX[K2 * NT];

  const int tid   = threadIdx.x;
  const int n0    = blockIdx.x * NT;
  const int b     = n0 >> 14;
  const int rem   = n0 & (HW - 1);
  const int y     = rem >> 7;
  const int x0col = rem & (WW - 1);

  // Phase 0: sampling coordinates for 9 taps x 64 pixels
  for (int e = tid; e < K2 * NT; e += 256) {
    int k  = e >> 6;
    int j  = e & 63;
    int xc = x0col + j;
    float dy = off[((size_t)(b * 18 + 2 * k) << 14) + (y << 7) + xc];
    float dx = off[((size_t)(b * 18 + 2 * k + 1) << 14) + (y << 7) + xc];
    float py = dy + (float)(k / 3 - 1) + (float)y;
    float px = dx + (float)(k % 3 - 1) + (float)xc;
    float fy = floorf(py), fx = floorf(px);
    sY0[e] = (int)fy;  sX0[e] = (int)fx;
    sWY[e] = py - fy;  sWX[e] = px - fx;
  }

  const int wave  = tid >> 5;
  const int lane  = tid & 31;
  const int otile = wave >> 1;     // 0..3
  const int npair = wave & 1;      // 0..1 -> n-tiles {2*npair, 2*npair+1}
  const int h     = lane >> 4;
  const int l15   = lane & 15;

  v8f c0 = {}, c1 = {};

  for (int cc = 0; cc < 2; ++cc) {
    __syncthreads();
    // Build packed patch chunk: 144 pair-rows x 64 cols
    for (int e = tid; e < CHUNK_P * NT; e += 256) {
      int pr = e >> 6;
      int j  = e & 63;
      unsigned pk = 0u;
      #pragma unroll
      for (int hv = 0; hv < 2; ++hv) {
        int Kg = cc * CHUNK_K + 2 * pr + hv;   // global K = c*9 + k
        int c  = Kg / 9;
        int k  = Kg - 9 * c;
        int ci = k * NT + j;
        int iy0 = sY0[ci], ix0 = sX0[ci];
        float wy = sWY[ci], wx = sWX[ci];
        const float* xb = x + ((size_t)(b * CC + c) << 14);
        int iy1 = iy0 + 1, ix1 = ix0 + 1;
        bool vy0 = (iy0 >= 0) & (iy0 < HH), vy1 = (iy1 >= 0) & (iy1 < HH);
        bool vx0 = (ix0 >= 0) & (ix0 < WW), vx1 = (ix1 >= 0) & (ix1 < WW);
        int cy0 = iy0 < 0 ? 0 : (iy0 > HH - 1 ? HH - 1 : iy0);
        int cy1 = iy1 < 0 ? 0 : (iy1 > HH - 1 ? HH - 1 : iy1);
        int cx0 = ix0 < 0 ? 0 : (ix0 > WW - 1 ? WW - 1 : ix0);
        int cx1 = ix1 < 0 ? 0 : (ix1 > WW - 1 ? WW - 1 : ix1);
        float v00 = (vy0 && vx0) ? xb[(cy0 << 7) + cx0] : 0.0f;
        float v01 = (vy0 && vx1) ? xb[(cy0 << 7) + cx1] : 0.0f;
        float v10 = (vy1 && vx0) ? xb[(cy1 << 7) + cx0] : 0.0f;
        float v11 = (vy1 && vx1) ? xb[(cy1 << 7) + cx1] : 0.0f;
        float s = (1.0f - wy) * ((1.0f - wx) * v00 + wx * v01)
                +          wy * ((1.0f - wx) * v10 + wx * v11);
        _Float16 hf = (_Float16)s;
        pk |= ((unsigned)__builtin_bit_cast(unsigned short, hf)) << (16 * hv);
      }
      Ppk[pr * PSTRIDE + j] = pk;
    }
    __syncthreads();

    // WMMA sweep: 9 steps of K=32; one A fragment feeds two B tiles
    const unsigned* wbase = wpk + (size_t)(otile * 16 + l15) * (KD / 2) + cc * CHUNK_P;
    const int bcol0 = npair * 32 + l15;
    for (int kk = 0; kk < CHUNK_P; kk += 16) {
      union { v16h v; unsigned u[8]; } A, B0, B1;
      #pragma unroll
      for (int r = 0; r < 8; ++r) {
        int apair = kk + h * 8 + r;
        A.u[r] = wbase[apair];
        int brow = (kk + h * 8 + r) * PSTRIDE;
        B0.u[r] = Ppk[brow + bcol0];
        B1.u[r] = Ppk[brow + bcol0 + 16];
      }
      c0 = __builtin_amdgcn_wmma_f32_16x16x32_f16(
          false, A.v, false, B0.v, (short)0, c0, false, false);
      c1 = __builtin_amdgcn_wmma_f32_16x16x32_f16(
          false, A.v, false, B1.v, (short)0, c1, false, false);
    }
  }

  // Epilogue: lane (h,l15) holds rows M = h*8 + r, col N = l15 within each tile
  const int pxb = x0col + npair * 32 + l15;
  float* orow0 = out + ((size_t)b << 20) + (y << 7) + pxb;
  #pragma unroll
  for (int r = 0; r < 8; ++r) {
    int o = otile * 16 + h * 8 + r;
    float bias = bd[o];
    orow0[(size_t)o << 14]      = c0[r] + bias;
    orow0[((size_t)o << 14) + 16] = c1[r] + bias;
  }
}

// ---------------------------------------------------------------------------
// 4) BN statistics: one block per channel -> scale/shift
// ---------------------------------------------------------------------------
__global__ __launch_bounds__(256) void bn_stats(const float* __restrict__ out,
                                                const float* __restrict__ gamma,
                                                const float* __restrict__ beta,
                                                float* __restrict__ scale,
                                                float* __restrict__ shift) {
  const int o = blockIdx.x;
  float s = 0.0f, s2 = 0.0f;
  for (int i = threadIdx.x; i < NPIX; i += 256) {
    int b = i >> 14, p = i & (HW - 1);
    float v = out[((size_t)b << 20) + ((size_t)o << 14) + p];
    s += v;
    s2 += v * v;
  }
  __shared__ float rs[256], rq[256];
  rs[threadIdx.x] = s;  rq[threadIdx.x] = s2;
  __syncthreads();
  for (int st = 128; st > 0; st >>= 1) {
    if (threadIdx.x < st) {
      rs[threadIdx.x] += rs[threadIdx.x + st];
      rq[threadIdx.x] += rq[threadIdx.x + st];
    }
    __syncthreads();
  }
  if (threadIdx.x == 0) {
    const float inv = 1.0f / (float)NPIX;
    float mean = rs[0] * inv;
    float var  = rq[0] * inv - mean * mean;
    float sc   = gamma[o] * rsqrtf(var + 1e-5f);
    scale[o] = sc;
    shift[o] = beta[o] - mean * sc;
  }
}

// ---------------------------------------------------------------------------
// 5) BN apply + ReLU, in place on d_out
// ---------------------------------------------------------------------------
__global__ __launch_bounds__(256) void bn_apply(float* __restrict__ out,
                                                const float* __restrict__ scale,
                                                const float* __restrict__ shift) {
  size_t i = (size_t)blockIdx.x * 256 + threadIdx.x;
  int o = (int)((i >> 14) & (OO - 1));
  float v = out[i] * scale[o] + shift[o];
  out[i] = v > 0.0f ? v : 0.0f;
}

// ---------------------------------------------------------------------------
extern "C" void kernel_launch(void* const* d_in, const int* in_sizes, int n_in,
                              void* d_out, int out_size, void* d_ws, size_t ws_size,
                              hipStream_t stream) {
  const float* x     = (const float*)d_in[0];
  const float* w_off = (const float*)d_in[1];
  const float* b_off = (const float*)d_in[2];
  const float* w_dcn = (const float*)d_in[3];
  const float* b_dcn = (const float*)d_in[4];
  const float* gamma = (const float*)d_in[5];
  const float* beta  = (const float*)d_in[6];
  float* out = (float*)d_out;

  // Workspace layout (bytes):
  //   [0)          offsets fp32 [4][18][128][128]       = 4,718,592
  //   [4,718,592)  w_dcn f16 [64][576]                  =    73,728
  //   [4,792,320)  w_off f16 [32][576] (rows>=18 zero)  =    36,864
  //   [4,829,184)  scale[64] fp32
  //   [4,829,440)  shift[64] fp32
  char* ws = (char*)d_ws;
  float*          off_buf  = (float*)ws;
  unsigned short* wf16_dcn = (unsigned short*)(ws + 4718592);
  unsigned short* wf16_off = (unsigned short*)(ws + 4792320);
  float*          scale    = (float*)(ws + 4829184);
  float*          shift    = (float*)(ws + 4829440);

  // 1) weight conversions (max 36864 elems)
  cvt_weights<<<(OO * KD + 255) / 256, 256, 0, stream>>>(w_dcn, w_off, wf16_dcn, wf16_off);
  // 2) offset conv on the matrix pipe
  offset_conv_wmma<<<NPIX / NT, 256, 0, stream>>>(x, (const unsigned*)wf16_off, b_off, off_buf);
  // 3) fused deformable sample + WMMA GEMM (+ bias), pre-BN into d_out
  dcn_main<<<NPIX / NT, 256, 0, stream>>>(x, off_buf, (const unsigned*)wf16_dcn, b_dcn, out);
  // 4) per-channel BN stats -> scale/shift
  bn_stats<<<OO, 256, 0, stream>>>(out, gamma, beta, scale, shift);
  // 5) normalize + ReLU in place
  bn_apply<<<(BB * OO * HW) / 256, 256, 0, stream>>>(out, scale, shift);
}
